// MemoryLocal_62672162784017
// MI455X (gfx1250) — matrix operand order repaired and checked by reading, not compiled, not run on hardware
//
#include <hip/hip_runtime.h>
#include <hip/hip_bf16.h>

typedef __attribute__((ext_vector_type(16))) _Float16 v16h;
typedef __attribute__((ext_vector_type(8)))  float    v8f;

#define B_  4
#define DE  64
#define T_  4
#define H_  128
#define W_  128
#define P2_ 49
#define HW  (H_*W_)
#define THW (T_*H_*W_)

#define LSTRIDE 212   // logit row stride: 848B (16B-aligned); 20i mod 64 distinct, i<16
#define LPAD0   196   // pad cols [196,208) = -1e30 so softmax scans a uniform 208 range
#define LTRASH  208   // trash col for banded extraction (outside scan range)
#define VSTRIDE 36    // lds_v row stride: 144B (16B-aligned), conflict-free

__global__ __launch_bounds__(32)
void MemoryLocal_62672162784017_kernel(const float* __restrict__ m_in,
                                       const float* __restrict__ m_out,
                                       const float* __restrict__ q_in,
                                       const float* __restrict__ cst_p,
                                       float* __restrict__ mem_out,
                                       float* __restrict__ attn_out)
{
  const int lane = threadIdx.x;      // 0..31 (wave32)
  const int half = lane >> 4;        // 0 or 1
  const int l16  = lane & 15;

  const int gid = blockIdx.x;        // B*H*(W/16) = 4096 strips
  const int w0  = (gid & 7) << 4;
  const int h   = (gid >> 3) & (H_ - 1);
  const int b   = gid >> 10;

  __shared__ float lds_logit[16][LSTRIDE]; // [pixel][t*49+dy*7+dx]
  __shared__ float lds_v[64][VSTRIDE];     // V-tile staging / mem-tile staging
  __shared__ float lds_red[16][4];         // softmax partials; [p][0] = 1/Z at end

  const float cst = cst_p[0];

  // Initialize softmax pad columns [196,208) to -1e30 (16 rows x 12 cols, 6/lane).
  {
    const int row = lane >> 1;
    const int c0  = LPAD0 + (lane & 1) * 6;
    #pragma unroll
    for (int j = 0; j < 6; ++j) lds_logit[row][c0 + j] = -1e30f;
  }

  // Per-strip, per-lane boundary predicates (hoisted; branch-free selects later).
  const int  wk0  = w0 - 3 + l16;            // phase-1 B key col, tile 0
  const int  wk1  = wk0 + 16;                // tile 1
  const bool ok0  = (wk0 >= 0);              // max col = 124 < 128, only low check
  const bool ok1  = (wk1 < W_);              // min col = 13 >= 0, only high check
  const int  wk0c = ok0 ? wk0 : 0;           // clamped (always-legal) addresses
  const int  wk1c = ok1 ? wk1 : (W_ - 1);

  const int  wp   = w0 - 3 + lane;           // phase-2 staging col (lane spans 32)
  const bool wok  = (wp >= 0) && (wp < W_);
  const int  wpc  = wok ? wp : 0;

  // ---- Query A fragments (16 pixels x 64 ch), kept in VGPRs for phase 1 ----
  // A 16x32 f16 layout: M = l16; element e -> K = half*16+e
  v16h aq0, aq1;
  {
    const float* qb = q_in + (size_t)b*DE*HW + (size_t)h*W_ + (w0 + l16);
    float t0[16], t1[16];
    #pragma unroll
    for (int e = 0; e < 16; ++e) {
      const int k = half*16 + e;
      t0[e] = qb[(size_t)(k)      * HW];
      t1[e] = qb[(size_t)(32 + k) * HW];
    }
    #pragma unroll
    for (int e = 0; e < 16; ++e) { aq0[e] = (_Float16)t0[e]; aq1[e] = (_Float16)t1[e]; }
  }

  const v8f vzero = {0.f,0.f,0.f,0.f,0.f,0.f,0.f,0.f};

  // =================== Phase 1: correlation logits ===================
  #pragma unroll 1
  for (int t = 0; t < T_; ++t) {
    #pragma unroll 1
    for (int dy = 0; dy < 7; ++dy) {
      const int r = h + dy - 3;                 // key row (workgroup-uniform)
      v8f acc0 = vzero, acc1 = vzero;           // S tiles: keys [0,16) and [16,32)
      if (r >= 0 && r < H_) {
        const float* kb = m_in + ((size_t)b*DE*T_ + t)*(size_t)HW + (size_t)r*W_;
        #pragma unroll
        for (int kc = 0; kc < 2; ++kc) {        // channel chunks of 32
          float t0[16], t1[16];                 // batch 32 unconditional loads
          #pragma unroll
          for (int e = 0; e < 16; ++e) {
            const int c = kc*32 + half*16 + e;  // B layout: K = half*16+e
            const float* cp = kb + (size_t)c * THW;
            t0[e] = cp[wk0c];
            t1[e] = cp[wk1c];
          }
          v16h b0, b1;
          #pragma unroll
          for (int e = 0; e < 16; ++e) {
            b0[e] = ok0 ? (_Float16)t0[e] : (_Float16)0.f;
            b1[e] = ok1 ? (_Float16)t1[e] : (_Float16)0.f;
          }
          const v16h a = kc ? aq1 : aq0;
          acc0 = __builtin_amdgcn_wmma_f32_16x16x32_f16(false, a, false, b0, (short)0, acc0, false, false);
          acc1 = __builtin_amdgcn_wmma_f32_16x16x32_f16(false, a, false, b1, (short)0, acc1, false, false);
        }
      }
      // Banded extraction S[m, m+dx] -> LDS; invalid entries go to trash col.
      // D layout: N = l16, VGPR rr -> M = rr + 8*half.
      const int base = t*P2_ + dy*7;
      #pragma unroll
      for (int rr = 0; rr < 8; ++rr) {
        const int m   = rr + half*8;
        const int dx0 = l16 - m;
        const int col0 = (dx0 >= 0 && dx0 <= 6) ? (base + dx0) : LTRASH;
        lds_logit[m][col0] = acc0[rr];
        const int dx1 = dx0 + 16;
        const int col1 = (dx1 <= 6) ? (base + dx1) : LTRASH;  // dx1 >= 1 always
        lds_logit[m][col1] = acc1[rr];
      }
    }
  }
  __syncthreads();

  // ======== Softmax over 196 (+const) channels (scan padded 208 range) ========
  {
    const int p   = l16;
    const float4* rowq = (const float4*)&lds_logit[p][0];
    float4*       roww = (float4*)&lds_logit[p][0];
    const int ch0 = half * 26;                 // 26 float4 chunks per half (104 floats)

    float mx = -1e30f;
    #pragma unroll
    for (int j = 0; j < 13; ++j) {             // 2 b128 loads per step, batched
      const float4 a = rowq[ch0 + 2*j];
      const float4 bq = rowq[ch0 + 2*j + 1];
      const float m0 = fmaxf(fmaxf(a.x,  a.y ), fmaxf(a.z,  a.w ));
      const float m1 = fmaxf(fmaxf(bq.x, bq.y), fmaxf(bq.z, bq.w));
      mx = fmaxf(mx, fmaxf(m0, m1));           // v_max_num_f32 tree
    }
    lds_red[p][half] = mx;
    __syncthreads();
    mx = fmaxf(fmaxf(lds_red[p][0], lds_red[p][1]), cst);
    float s0 = 0.f, s1 = 0.f;
    #pragma unroll
    for (int j = 0; j < 13; ++j) {
      const float4 a = rowq[ch0 + 2*j];
      const float4 bq = rowq[ch0 + 2*j + 1];
      float4 ea, eb;
      ea.x = __expf(a.x  - mx); ea.y = __expf(a.y  - mx);
      ea.z = __expf(a.z  - mx); ea.w = __expf(a.w  - mx);
      eb.x = __expf(bq.x - mx); eb.y = __expf(bq.y - mx);
      eb.z = __expf(bq.z - mx); eb.w = __expf(bq.w - mx);
      roww[ch0 + 2*j]     = ea;                // pads become exp(-huge)=0
      roww[ch0 + 2*j + 1] = eb;
      s0 += (ea.x + ea.y) + (ea.z + ea.w);
      s1 += (eb.x + eb.y) + (eb.z + eb.w);
    }
    lds_red[p][2 + half] = s0 + s1;
    __syncthreads();
    const float s   = lds_red[p][2] + lds_red[p][3] + __expf(cst - mx);
    const float inv = 1.f / s;
    lds_red[p][0] = inv;                       // both halves write same value
    // attn output: lane's two t values; multiply by inv on the fly.
    float* ap = attn_out + (size_t)b*P2_*T_*HW + (size_t)h*W_ + (w0 + p);
    #pragma unroll
    for (int tt = 0; tt < 2; ++tt) {
      const int t = half*2 + tt;
      float* apt = ap + (size_t)t*HW;
      #pragma unroll 7
      for (int d = 0; d < P2_; ++d) {
        apt[(size_t)d*T_*HW] = lds_logit[p][t*P2_ + d] * inv;
      }
    }
  }
  __syncthreads();

  // ============ Phase 2: attention read (banded GEMM, unnormalized) ============
  v8f macc[4] = {vzero, vzero, vzero, vzero};  // mem[16 pixels][64 ch] in 4 N-tiles
  #pragma unroll 1
  for (int t = 0; t < T_; ++t) {
    #pragma unroll 1
    for (int dy = 0; dy < 7; ++dy) {
      const int r = h + dy - 3;
      if (r < 0 || r >= H_) continue;          // uniform: zero-padded rows drop out
      const float* vb = m_out + ((size_t)b*DE*T_ + t)*(size_t)HW + (size_t)r*W_
                              + wpc;           // per-lane clamped key column
      __syncthreads();                         // protect lds_v from prior readers
      #pragma unroll
      for (int cc = 0; cc < 64; cc += 16) {    // stage V tile [64 ch][32 keys]
        float tmp[16];
        #pragma unroll
        for (int j = 0; j < 16; ++j) tmp[j] = vb[(size_t)(cc + j) * THW];
        #pragma unroll
        for (int j = 0; j < 16; ++j) lds_v[cc + j][lane] = wok ? tmp[j] : 0.f;
      }
      __syncthreads();
      // A fragment: banded unnormalized attn A[m,k] = e[m][dy][k-m], 0<=k-m<=6.
      const int base = t*P2_ + dy*7;
      v16h af;
      #pragma unroll
      for (int e = 0; e < 16; ++e) {
        const int k   = half*16 + e;
        const int dx  = k - l16;               // M = l16
        const bool in = (dx >= 0) && (dx <= 6);
        const int dxc = in ? dx : 0;           // clamped, unconditional ds load
        const float v = lds_logit[l16][base + dxc];
        af[e] = in ? (_Float16)v : (_Float16)0.f;
      }
      // B fragments from LDS via float4 (ds_load_b128): N = l16, K = half*16+e.
      #pragma unroll
      for (int nc = 0; nc < 4; ++nc) {
        const int c = nc*16 + l16;
        const float4* rowp = (const float4*)&lds_v[c][half*16];
        float4 q0 = rowp[0], q1 = rowp[1], q2 = rowp[2], q3 = rowp[3];
        v16h bv;
        bv[0]  = (_Float16)q0.x; bv[1]  = (_Float16)q0.y;
        bv[2]  = (_Float16)q0.z; bv[3]  = (_Float16)q0.w;
        bv[4]  = (_Float16)q1.x; bv[5]  = (_Float16)q1.y;
        bv[6]  = (_Float16)q1.z; bv[7]  = (_Float16)q1.w;
        bv[8]  = (_Float16)q2.x; bv[9]  = (_Float16)q2.y;
        bv[10] = (_Float16)q2.z; bv[11] = (_Float16)q2.w;
        bv[12] = (_Float16)q3.x; bv[13] = (_Float16)q3.y;
        bv[14] = (_Float16)q3.z; bv[15] = (_Float16)q3.w;
        macc[nc] = __builtin_amdgcn_wmma_f32_16x16x32_f16(false, af, false, bv, (short)0, macc[nc], false, false);
      }
    }
  }

  // ---- Apply per-pixel 1/Z to D rows, then stage + b128 writeback ----
  float invv[8];
  #pragma unroll
  for (int rr = 0; rr < 8; ++rr) invv[rr] = lds_red[rr + 8*half][0];
  __syncthreads();
  #pragma unroll
  for (int nc = 0; nc < 4; ++nc) {
    const int c = nc*16 + l16;                 // D: N = l16 -> channel
    #pragma unroll
    for (int rr = 0; rr < 8; ++rr) {
      lds_v[c][rr + half*8] = macc[nc][rr] * invv[rr];  // D: VGPR rr -> pixel m
    }
  }
  __syncthreads();
  float* ob = mem_out + ((size_t)b*DE*H_ + h)*W_ + w0;
  #pragma unroll
  for (int it = 0; it < 8; ++it) {             // 256 float4 chunks / 32 lanes
    const int chunk = it*32 + lane;
    const int c  = chunk >> 2;
    const int m4 = (chunk & 3) << 2;
    const float4 v = *(const float4*)&lds_v[c][m4];
    *(float4*)&ob[(size_t)c*HW + m4] = v;
  }
}

extern "C" void kernel_launch(void* const* d_in, const int* in_sizes, int n_in,
                              void* d_out, int out_size, void* d_ws, size_t ws_size,
                              hipStream_t stream) {
  (void)in_sizes; (void)n_in; (void)out_size; (void)d_ws; (void)ws_size;
  const float* m_in  = (const float*)d_in[0];
  const float* m_out = (const float*)d_in[1];
  const float* q_in  = (const float*)d_in[2];
  const float* cst   = (const float*)d_in[3];
  float* mem  = (float*)d_out;
  float* attn = (float*)d_out + (size_t)B_*DE*HW;   // attn follows mem, flat
  dim3 grid(B_ * H_ * (W_ / 16));
  dim3 block(32);
  MemoryLocal_62672162784017_kernel<<<grid, block, 0, stream>>>(m_in, m_out, q_in, cst, mem, attn);
}
